// RelationMultiHeadSelfAttention_25761213841786
// MI455X (gfx1250) — matrix operand order repaired and checked by reading, not compiled, not run on hardware
//
#include <hip/hip_runtime.h>

typedef _Float16 v16h __attribute__((ext_vector_type(16)));
typedef float    v8f  __attribute__((ext_vector_type(8)));

#define KT_P 264   // LDS pitch (halves) for k-transpose tile
#define AT_P 272   // LDS pitch (halves) for attn tile

// ---------------------------------------------------------------------------
// Kernel 1: fold fc_fuse + projections into M (32x32), wr_eff (32), bsum (1)
// ws layout (floats): [0,1024) M ; [1024,1056) wr_eff ; [1056] bsum
// ---------------------------------------------------------------------------
__global__ __launch_bounds__(256) void prep_kernel(const float* __restrict__ Wq,
                                                   const float* __restrict__ Wk,
                                                   const float* __restrict__ Wr,
                                                   const float* __restrict__ Wfuse,
                                                   const float* __restrict__ bfuse,
                                                   float* __restrict__ ws) {
  __shared__ float s_wsum[64];
  const int tid = threadIdx.x;
  if (tid < 64) {
    float s = 0.f;
    for (int d = 0; d < 32; ++d) s += Wfuse[d * 64 + tid];
    s_wsum[tid] = s;
  }
  __syncthreads();
  // M[d][dp] = sum_o Wq[o,d] * wsum[o] * Wk[o,dp]
  for (int e = tid; e < 1024; e += 256) {
    const int d = e >> 5, dp = e & 31;
    float m = 0.f;
    for (int o = 0; o < 32; ++o) m += Wq[o * 32 + d] * s_wsum[o] * Wk[o * 32 + dp];
    ws[e] = m;
  }
  if (tid < 32) {
    float s = 0.f;
    for (int o = 0; o < 32; ++o) s += s_wsum[32 + o] * Wr[o * 32 + tid];
    ws[1024 + tid] = s;
  }
  if (tid == 0) {
    float s = 0.f;
    for (int d = 0; d < 32; ++d) s += bfuse[d];
    ws[1056] = s;
  }
}

// ---------------------------------------------------------------------------
// Kernel 2: qm[n][h][q][dp] = (f16) sum_d query[n,q,h*32+d] * M[d][dp]
//           vp[n][h][k][o]  = (f16) sum_d values[n,k,h*32+d] * Wv[o][d]
// ---------------------------------------------------------------------------
__global__ __launch_bounds__(256) void project_kernel(const float* __restrict__ query,
                                                      const float* __restrict__ values,
                                                      const float* __restrict__ Mw,
                                                      const float* __restrict__ Wv,
                                                      _Float16* __restrict__ qm_h,
                                                      _Float16* __restrict__ vp_h) {
  const int i  = blockIdx.x * 256 + threadIdx.x;   // ((n*8+h)*256+q)*32+dp
  const int dp = i & 31;
  const int q  = (i >> 5) & 255;
  const int h  = (i >> 13) & 7;
  const int n  = i >> 16;
  const size_t src = ((size_t)(n * 256 + q)) * 256 + h * 32;
  if (blockIdx.y == 0) {
    const float* x = query + src;
    float acc = 0.f;
    #pragma unroll
    for (int d = 0; d < 32; ++d) acc += x[d] * Mw[d * 32 + dp];
    qm_h[i] = (_Float16)acc;
  } else {
    const float* x = values + src;
    float acc = 0.f;
    #pragma unroll
    for (int d = 0; d < 32; ++d) acc += x[d] * Wv[dp * 32 + d];
    vp_h[i] = (_Float16)acc;
  }
}

// ---------------------------------------------------------------------------
// Kernel 3: fused attention per (n, h, 16-row q-tile). 256 threads = 8 waves.
// ---------------------------------------------------------------------------
__global__ __launch_bounds__(256) void attn_kernel(const float* __restrict__ keys,
                                                   const int* __restrict__ mask,
                                                   const float* __restrict__ relation,
                                                   const float* __restrict__ wr,
                                                   const float* __restrict__ bsum_p,
                                                   const _Float16* __restrict__ qm_h,
                                                   const _Float16* __restrict__ vp_h,
                                                   float* __restrict__ outH) {
  __shared__ _Float16 s_kT[32 * KT_P];     // kT[d][k]  (f16)
  __shared__ float    s_energy[16 * 256];  // energy[qrow][k]
  __shared__ _Float16 s_attn[16 * AT_P];   // attn[qrow][k]  (f16)
  __shared__ float    s_part[8 * 16 * 16]; // per-wave partial C tiles

  const int qt = blockIdx.x, h = blockIdx.y, n = blockIdx.z;
  const int q0 = qt * 16;
  const int tid = threadIdx.x;
  const int lane = tid & 31, wv = tid >> 5;
  const int half_ = lane >> 4, lq = lane & 15;

  // ---- stage kT[d][k] = keys[n,k,h*32+d] as f16 ----
  {
    const int k = tid;
    const float* kp = keys + ((size_t)(n * 256 + k)) * 256 + h * 32;
    #pragma unroll
    for (int d = 0; d < 32; ++d) s_kT[d * KT_P + k] = (_Float16)kp[d];
  }

  // wr_eff into registers (float4 x 8)
  float4 w4[8];
  const float4* wrp = (const float4*)wr;
  #pragma unroll
  for (int u = 0; u < 8; ++u) w4[u] = wrp[u];
  const float bsumv = bsum_p[0];

  // ---- A-matrix (16x32 f16): qm tile, lane = q-row, loop-invariant ----
  v16h a;
  {
    const _Float16* qp = qm_h + ((size_t)((n * 8 + h) * 256 + q0) + lq) * 32;
    #pragma unroll
    for (int j = 0; j < 8; ++j) {
      const int Kb = 2 * j + 8 * half_ + (j >= 4 ? 8 : 0);
      a[2 * j]     = qp[Kb];
      a[2 * j + 1] = qp[Kb + 1];
    }
  }
  __syncthreads();

  // ---- Phase A: energy = qM.k (WMMA) + relation.wr_eff + bsum ----
  for (int tt = 0; tt < 2; ++tt) {
    const int kb = (wv + tt * 8) * 16;
    v16h b;
    #pragma unroll
    for (int j = 0; j < 8; ++j) {
      const int Kd = 2 * j + 8 * half_ + (j >= 4 ? 8 : 0);
      b[2 * j]     = s_kT[Kd * KT_P + kb + lq];
      b[2 * j + 1] = s_kT[(Kd + 1) * KT_P + kb + lq];
    }
    v8f c = {0.f, 0.f, 0.f, 0.f, 0.f, 0.f, 0.f, 0.f};
    c = __builtin_amdgcn_wmma_f32_16x16x32_f16(false, a, false, b, (short)0, c, false, false);

    const int k = kb + lq;
    #pragma unroll
    for (int i = 0; i < 8; ++i) {
      const int qrow = i + 8 * half_;                 // C layout: M = i + 8*half
      const float4* rp = (const float4*)(relation +
          (((size_t)(n * 256 + q0 + qrow)) * 256 + k) * 256 + h * 32);
      float acc = c[i] + bsumv;
      #pragma unroll
      for (int u = 0; u < 8; ++u) {
        const float4 rv = rp[u];
        acc += rv.x * w4[u].x + rv.y * w4[u].y + rv.z * w4[u].z + rv.w * w4[u].w;
      }
      s_energy[qrow * 256 + k] = acc;
    }
  }
  __syncthreads();

  // ---- softmax over k (mask, scale 1/sqrt(256)=1/16), rows 2*wv, 2*wv+1 ----
  for (int rr = 0; rr < 2; ++rr) {
    const int r = 2 * wv + rr;
    const int q = q0 + r;
    const int* mp = mask + ((size_t)n * 256 + q) * 256;
    float ev[8];
    float vmax = -3.0e38f;
    #pragma unroll
    for (int u = 0; u < 8; ++u) {
      const int kk = lane + 32 * u;
      float e = s_energy[r * 256 + kk];
      if (mp[kk] == 1) e = -1.0e20f;
      e *= 0.0625f;
      ev[u] = e;
      vmax = fmaxf(vmax, e);
    }
    #pragma unroll
    for (int off = 16; off > 0; off >>= 1) vmax = fmaxf(vmax, __shfl_xor(vmax, off, 32));
    float sum = 0.f;
    #pragma unroll
    for (int u = 0; u < 8; ++u) { ev[u] = __expf(ev[u] - vmax); sum += ev[u]; }
    #pragma unroll
    for (int off = 16; off > 0; off >>= 1) sum += __shfl_xor(sum, off, 32);
    const float inv = 1.f / sum;
    #pragma unroll
    for (int u = 0; u < 8; ++u) s_attn[r * AT_P + lane + 32 * u] = (_Float16)(ev[u] * inv);
  }
  __syncthreads();

  // ---- Phase C: out16x32 = attn(16x256) @ vproj(256x32) via WMMA ----
  const int dtile = wv >> 2;                      // waves 0-3: d 0..15, 4-7: d 16..31
  const size_t bnh = (size_t)(n * 8 + h) * 256;
  v8f c2 = {0.f, 0.f, 0.f, 0.f, 0.f, 0.f, 0.f, 0.f};
  for (int cc = 0; cc < 2; ++cc) {
    const int kb = ((wv & 3) * 2 + cc) * 32;      // 32-wide K chunk
    v16h a2, b2;
    #pragma unroll
    for (int j = 0; j < 8; ++j) {
      const int Kb = 2 * j + 8 * half_ + (j >= 4 ? 8 : 0);
      a2[2 * j]     = s_attn[lq * AT_P + kb + Kb];
      a2[2 * j + 1] = s_attn[lq * AT_P + kb + Kb + 1];
    }
    #pragma unroll
    for (int j = 0; j < 8; ++j) {
      const int Kd = 2 * j + 8 * half_ + (j >= 4 ? 8 : 0);
      const _Float16* vpp = vp_h + (bnh + kb + Kd) * 32 + dtile * 16 + lq;
      b2[2 * j]     = vpp[0];
      b2[2 * j + 1] = vpp[32];
    }
    c2 = __builtin_amdgcn_wmma_f32_16x16x32_f16(false, a2, false, b2, (short)0, c2, false, false);
  }
  #pragma unroll
  for (int i = 0; i < 8; ++i)
    s_part[(wv * 16 + i + 8 * half_) * 16 + lq] = c2[i];
  __syncthreads();

  // ---- reduce 4 partial waves per d-tile, write outH (N,Q,H,D) ----
  for (int rep = 0; rep < 2; ++rep) {
    const int idx = rep * 256 + tid;
    const int qrow = idx >> 5, d = idx & 31;
    const int dt = d >> 4, dn = d & 15;
    float s = 0.f;
    #pragma unroll
    for (int ww = 0; ww < 4; ++ww) s += s_part[((dt * 4 + ww) * 16 + qrow) * 16 + dn];
    outH[((size_t)(n * 256 + q0 + qrow)) * 256 + h * 32 + d] = s;
  }
}

// ---------------------------------------------------------------------------
// Kernel 4: out = outH @ Wout^T + bout  (row-stationary; Wout stays in L2)
// ---------------------------------------------------------------------------
__global__ __launch_bounds__(256) void outproj_kernel(const float* __restrict__ outH,
                                                      const float* __restrict__ Wout,
                                                      const float* __restrict__ bout,
                                                      float* __restrict__ out) {
  __shared__ float s_row[256];
  const int row = blockIdx.x, tid = threadIdx.x;
  s_row[tid] = outH[(size_t)row * 256 + tid];
  __syncthreads();
  const float4* wrow = (const float4*)(Wout + (size_t)tid * 256);
  const float4* sr = (const float4*)s_row;
  float acc = bout[tid];
  #pragma unroll 8
  for (int f = 0; f < 64; ++f) {
    const float4 wv_ = wrow[f], sv = sr[f];
    acc += wv_.x * sv.x + wv_.y * sv.y + wv_.z * sv.z + wv_.w * sv.w;
  }
  out[(size_t)row * 256 + tid] = acc;
}

// ---------------------------------------------------------------------------
extern "C" void kernel_launch(void* const* d_in, const int* in_sizes, int n_in,
                              void* d_out, int out_size, void* d_ws, size_t ws_size,
                              hipStream_t stream) {
  (void)in_sizes; (void)n_in; (void)out_size; (void)ws_size;
  const float* values   = (const float*)d_in[0];
  const float* keys     = (const float*)d_in[1];
  const float* query    = (const float*)d_in[2];
  const int*   mask     = (const int*)  d_in[3];
  const float* relation = (const float*)d_in[4];
  const float* Wv       = (const float*)d_in[5];
  const float* Wk       = (const float*)d_in[6];
  const float* Wq       = (const float*)d_in[7];
  const float* Wr       = (const float*)d_in[8];
  const float* Wfuse    = (const float*)d_in[9];
  const float* bfuse    = (const float*)d_in[10];
  const float* Wout     = (const float*)d_in[11];
  const float* bout     = (const float*)d_in[12];
  float* out = (float*)d_out;

  // workspace layout (floats)
  float* wsf   = (float*)d_ws;
  float* Mw    = wsf;                                   // 1024
  float* wr    = wsf + 1024;                            // 32
  float* bsum  = wsf + 1056;                            // 1 (+pad)
  _Float16* qm_h = (_Float16*)(wsf + 1088);             // 262144 halves
  _Float16* vp_h = (_Float16*)(wsf + 1088 + 131072);    // 262144 halves
  float* outH  = wsf + 1088 + 262144;                   // 262144 floats

  prep_kernel<<<1, 256, 0, stream>>>(Wq, Wk, Wr, Wfuse, bfuse, wsf);
  project_kernel<<<dim3(1024, 2), 256, 0, stream>>>(query, values, Mw, Wv, qm_h, vp_h);
  attn_kernel<<<dim3(16, 8, 4), 256, 0, stream>>>(keys, mask, relation, wr, bsum, qm_h, vp_h, outH);
  outproj_kernel<<<1024, 256, 0, stream>>>(outH, Wout, bout, out);
}
